// TermEncoder_16423954940619
// MI455X (gfx1250) — matrix operand order repaired and checked by reading, not compiled, not run on hardware
//
#include <hip/hip_runtime.h>
#include <hip/hip_bf16.h>

#define NODES 50000
#define EDGES 800000
#define DIM 128
#define HEADS 2
#define CH 128
#define HC 256
#define NEG_SLOPE 0.2f
#define GAT_EPS 1e-16f

typedef __attribute__((ext_vector_type(16))) __bf16 v16bf;
typedef __attribute__((ext_vector_type(8)))  float  v8f;

union Frag {
  v16bf v;
  uint4 q[2];
};

__device__ __forceinline__ unsigned short f2bf(float f) {
  unsigned u = __float_as_uint(f);
  u += 0x7FFFu + ((u >> 16) & 1u);           // round-to-nearest-even
  return (unsigned short)(u >> 16);
}

// monotone float <-> uint mapping so unsigned atomicMax == float max
__device__ __forceinline__ unsigned ordf(float f) {
  unsigned u = __float_as_uint(f);
  return (u & 0x80000000u) ? ~u : (u | 0x80000000u);
}
__device__ __forceinline__ float unordf(unsigned u) {
  unsigned v = (u & 0x80000000u) ? (u & 0x7FFFFFFFu) : ~u;
  return __uint_as_float(v);
}

// ---------------------------------------------------------------------------
// Tiled bf16 WMMA GEMM:  out[M x NOUT] = X[M x K] * W[NOUT x K]^T (+bias)
// One wave computes a 16-row band; A fragments held in registers for all K.
// MODE 0: f32 out, no bias.  MODE 1: bias + bf16 out.  MODE 2: bias + f32 out.
// ---------------------------------------------------------------------------
template <int K, int NOUT, int MODE>
__global__ __launch_bounds__(256) void gat_gemm_kernel(
    const unsigned short* __restrict__ X,
    const unsigned short* __restrict__ W,
    const float* __restrict__ bias,
    float* __restrict__ outF,
    unsigned short* __restrict__ outB,
    int M)
{
  constexpr int KB = K / 32;
  const int lane = threadIdx.x & 31;
  const int wave = blockIdx.x * (blockDim.x >> 5) + (threadIdx.x >> 5);
  const int row0 = wave * 16;
  if (row0 >= M) return;                      // wave-uniform: EXEC all-ones for WMMA
  const int mrow = lane & 15;
  const int hi   = lane >> 4;

  // A fragment per ISA 16-bit A layout: lanes<16 -> K {0..7,16..23}, lanes>=16 -> {8..15,24..31}
  Frag a[KB];
  const unsigned short* xrow = X + (size_t)(row0 + mrow) * K;
#pragma unroll
  for (int kb = 0; kb < KB; ++kb) {
    a[kb].q[0] = *(const uint4*)(xrow + kb * 32 + hi * 8);
    a[kb].q[1] = *(const uint4*)(xrow + kb * 32 + 16 + hi * 8);
  }

  for (int nt = 0; nt < NOUT / 16; ++nt) {
    v8f acc = {0.f, 0.f, 0.f, 0.f, 0.f, 0.f, 0.f, 0.f};
    // B = W^T; lane holds column n = lane&15, K {hi*8..+7, 16+hi*8..+7} (matches A layout)
    const unsigned short* wrow = W + (size_t)(nt * 16 + mrow) * K;
#pragma unroll
    for (int kb = 0; kb < KB; ++kb) {
      Frag b;
      b.q[0] = *(const uint4*)(wrow + kb * 32 + hi * 8);
      b.q[1] = *(const uint4*)(wrow + kb * 32 + 16 + hi * 8);
      acc = __builtin_amdgcn_wmma_f32_16x16x32_bf16(
          false, a[kb].v, false, b.v, (short)0, acc, false, false);
    }
    const int col = nt * 16 + mrow;
    const float bv = (MODE != 0) ? bias[col] : 0.f;
#pragma unroll
    for (int r = 0; r < 8; ++r) {             // D layout: M = r + hi*8, N = lane&15
      const int m = row0 + r + hi * 8;
      const float v = acc[r] + bv;
      if (MODE == 1) outB[(size_t)m * NOUT + col] = f2bf(v);
      else           outF[(size_t)m * NOUT + col] = v;
    }
  }
}

// alpha_l/alpha_r: one wave per (node, head); 128-length dot, wave32 shuffle reduce
__global__ __launch_bounds__(256) void gat_alpha_kernel(
    const float* __restrict__ xl, const float* __restrict__ attl,
    const float* __restrict__ attr, float* __restrict__ al, float* __restrict__ ar)
{
  const int lane = threadIdx.x & 31;
  const int wave = blockIdx.x * 8 + (threadIdx.x >> 5);
  if (wave >= NODES * HEADS) return;
  const int n = wave >> 1, h = wave & 1;
  const float4 xv = *(const float4*)(xl + (size_t)n * HC + h * CH + lane * 4);
  const float4 lv = *(const float4*)(attl + h * CH + lane * 4);
  const float4 rv = *(const float4*)(attr + h * CH + lane * 4);
  float sl = xv.x * lv.x + xv.y * lv.y + xv.z * lv.z + xv.w * lv.w;
  float sr = xv.x * rv.x + xv.y * rv.y + xv.z * rv.z + xv.w * rv.w;
  for (int o = 16; o > 0; o >>= 1) {
    sl += __shfl_down(sl, o, 32);
    sr += __shfl_down(sr, o, 32);
  }
  if (lane == 0) { al[wave] = sl; ar[wave] = sr; }
}

__global__ void gat_node_init_kernel(unsigned* __restrict__ mU, float* __restrict__ denom)
{
  const int i = blockIdx.x * blockDim.x + threadIdx.x;
  if (i < NODES * HEADS) { mU[i] = 0x007FFFFFu /* ordf(-inf) */; denom[i] = 0.f; }
}

__global__ void gat_zero_f32_kernel(float* __restrict__ p, int n)
{
  const int i = blockIdx.x * blockDim.x + threadIdx.x;
  if (i < n) p[i] = 0.f;
}

// pass A: per-edge leaky-relu logit -> segment max (ordered-uint atomicMax)
__global__ __launch_bounds__(256) void gat_edge_max_kernel(
    const int* __restrict__ ei, const float* __restrict__ al,
    const float* __restrict__ ar, unsigned* __restrict__ mU)
{
  const int e = blockIdx.x * blockDim.x + threadIdx.x;
  if (e >= EDGES) return;
  const int s = ei[e], d = ei[EDGES + e];
#pragma unroll
  for (int h = 0; h < HEADS; ++h) {
    float v = al[s * HEADS + h] + ar[d * HEADS + h];
    v = v > 0.f ? v : NEG_SLOPE * v;
    atomicMax(&mU[d * HEADS + h], ordf(v));
  }
}

// pass B: ex = exp(e - m), denom += ex
__global__ __launch_bounds__(256) void gat_edge_exp_kernel(
    const int* __restrict__ ei, const float* __restrict__ al,
    const float* __restrict__ ar, const unsigned* __restrict__ mU,
    float* __restrict__ denom, float* __restrict__ ex)
{
  const int e = blockIdx.x * blockDim.x + threadIdx.x;
  if (e >= EDGES) return;
  const int s = ei[e], d = ei[EDGES + e];
#pragma unroll
  for (int h = 0; h < HEADS; ++h) {
    float v = al[s * HEADS + h] + ar[d * HEADS + h];
    v = v > 0.f ? v : NEG_SLOPE * v;
    float m = unordf(mU[d * HEADS + h]);
    if (!isfinite(m)) m = 0.f;                // matches jnp.where(isfinite(m), m, 0)
    const float x = __expf(v - m);
    ex[(size_t)e * HEADS + h] = x;
    atomicAdd(&denom[d * HEADS + h], x);
  }
}

// pass C: agg[dst] += a * xl[src]; one wave per edge, 8 channels per lane
__global__ __launch_bounds__(256) void gat_edge_agg_kernel(
    const int* __restrict__ ei, const float* __restrict__ ex,
    const float* __restrict__ denom, const float* __restrict__ xl,
    float* __restrict__ agg)
{
  const long long tid = (long long)blockIdx.x * blockDim.x + threadIdx.x;
  const int e = (int)(tid >> 5);
  if (e >= EDGES) return;
  const int lane = (int)(tid & 31);
  const int s = ei[e], d = ei[EDGES + e];
  const float a0 = ex[(size_t)e * 2 + 0] / (denom[d * 2 + 0] + GAT_EPS);
  const float a1 = ex[(size_t)e * 2 + 1] / (denom[d * 2 + 1] + GAT_EPS);
  const int c = lane * 8;
  const float aa = (c < CH) ? a0 : a1;        // 8-chunk never straddles head boundary
  const float4 x0 = *(const float4*)(xl + (size_t)s * HC + c);
  const float4 x1 = *(const float4*)(xl + (size_t)s * HC + c + 4);
  float* o = agg + (size_t)d * HC + c;
  atomicAdd(o + 0, aa * x0.x); atomicAdd(o + 1, aa * x0.y);
  atomicAdd(o + 2, aa * x0.z); atomicAdd(o + 3, aa * x0.w);
  atomicAdd(o + 4, aa * x1.x); atomicAdd(o + 5, aa * x1.y);
  atomicAdd(o + 6, aa * x1.z); atomicAdd(o + 7, aa * x1.w);
}

__global__ void gat_relu_bf_kernel(const float* __restrict__ in,
                                   unsigned short* __restrict__ out, int n)
{
  const int i = blockIdx.x * blockDim.x + threadIdx.x;
  if (i < n) { float v = in[i]; out[i] = f2bf(v > 0.f ? v : 0.f); }
}

__global__ void gat_f2bf_kernel(const float* __restrict__ in,
                                unsigned short* __restrict__ out, int n)
{
  const int i = blockIdx.x * blockDim.x + threadIdx.x;
  if (i < n) out[i] = f2bf(in[i]);
}

extern "C" void kernel_launch(void* const* d_in, const int* in_sizes, int n_in,
                              void* d_out, int out_size, void* d_ws, size_t ws_size,
                              hipStream_t stream)
{
  (void)in_sizes; (void)n_in; (void)out_size; (void)ws_size;
  const float* x     = (const float*)d_in[0];
  const int*   ei    = (const int*)d_in[1];   // jnp int64 -> int32 (x64 disabled)
  const float* W1    = (const float*)d_in[2];
  const float* attl1 = (const float*)d_in[3];
  const float* attr1 = (const float*)d_in[4];
  const float* W2    = (const float*)d_in[5];
  const float* attl2 = (const float*)d_in[6];
  const float* attr2 = (const float*)d_in[7];
  const float* Wp1   = (const float*)d_in[8];
  const float* bp1   = (const float*)d_in[9];
  const float* Wp2   = (const float*)d_in[10];
  const float* bp2   = (const float*)d_in[11];
  float* out = (float*)d_out;

  char* ws = (char*)d_ws;
  size_t off = 0;
  auto take = [&](size_t bytes) -> char* {
    char* p = ws + off;
    off += (bytes + 255) & ~(size_t)255;
    return p;
  };
  unsigned short* w1b   = (unsigned short*)take((size_t)HC * DIM * 2);
  unsigned short* w2b   = (unsigned short*)take((size_t)HC * HC * 2);
  unsigned short* wp1b  = (unsigned short*)take((size_t)CH * HC * 2);
  unsigned short* wp2b  = (unsigned short*)take((size_t)CH * CH * 2);
  unsigned short* featb = (unsigned short*)take((size_t)NODES * HC * 2);
  float*    xl    = (float*)take((size_t)NODES * HC * 4);
  float*    agg   = (float*)take((size_t)NODES * HC * 4);
  float*    al    = (float*)take((size_t)NODES * HEADS * 4);
  float*    ar    = (float*)take((size_t)NODES * HEADS * 4);
  unsigned* mU    = (unsigned*)take((size_t)NODES * HEADS * 4);
  float*    denom = (float*)take((size_t)NODES * HEADS * 4);
  float*    ex    = (float*)take((size_t)EDGES * HEADS * 4);
  unsigned short* p1b = (unsigned short*)xl;  // reuse: xl dead before proj1

  const dim3 blk(256);
  const int gemmBlocks = (NODES / 16 + 7) / 8;     // 8 waves/block, 16 rows/wave
  const int nhBlocks   = (NODES * HEADS + 255) / 256;
  const int ncBlocks   = ((int)((size_t)NODES * HC) + 255) / 256;
  const int eBlocks    = (EDGES + 255) / 256;
  const int aggBlocks  = (int)(((size_t)EDGES * 32 + 255) / 256);
  const int waveBlocks = (NODES * HEADS + 7) / 8;  // one wave per (n,h)

  // bf16 conversions (weights + input features)
  gat_f2bf_kernel<<<(HC * DIM + 255) / 256, blk, 0, stream>>>(W1, w1b, HC * DIM);
  gat_f2bf_kernel<<<(HC * HC + 255) / 256, blk, 0, stream>>>(W2, w2b, HC * HC);
  gat_f2bf_kernel<<<(CH * HC + 255) / 256, blk, 0, stream>>>(Wp1, wp1b, CH * HC);
  gat_f2bf_kernel<<<(CH * CH + 255) / 256, blk, 0, stream>>>(Wp2, wp2b, CH * CH);
  gat_f2bf_kernel<<<(NODES * DIM + 255) / 256, blk, 0, stream>>>(x, featb, NODES * DIM);

  // ---- GAT layer 1 ----
  gat_gemm_kernel<DIM, HC, 0><<<gemmBlocks, blk, 0, stream>>>(featb, w1b, nullptr, xl, nullptr, NODES);
  gat_alpha_kernel<<<waveBlocks, blk, 0, stream>>>(xl, attl1, attr1, al, ar);
  gat_node_init_kernel<<<nhBlocks, blk, 0, stream>>>(mU, denom);
  gat_zero_f32_kernel<<<ncBlocks, blk, 0, stream>>>(agg, NODES * HC);
  gat_edge_max_kernel<<<eBlocks, blk, 0, stream>>>(ei, al, ar, mU);
  gat_edge_exp_kernel<<<eBlocks, blk, 0, stream>>>(ei, al, ar, mU, denom, ex);
  gat_edge_agg_kernel<<<aggBlocks, blk, 0, stream>>>(ei, ex, denom, xl, agg);
  gat_relu_bf_kernel<<<ncBlocks, blk, 0, stream>>>(agg, featb, NODES * HC);

  // ---- GAT layer 2 ----
  gat_gemm_kernel<HC, HC, 0><<<gemmBlocks, blk, 0, stream>>>(featb, w2b, nullptr, xl, nullptr, NODES);
  gat_alpha_kernel<<<waveBlocks, blk, 0, stream>>>(xl, attl2, attr2, al, ar);
  gat_node_init_kernel<<<nhBlocks, blk, 0, stream>>>(mU, denom);
  gat_zero_f32_kernel<<<ncBlocks, blk, 0, stream>>>(agg, NODES * HC);
  gat_edge_max_kernel<<<eBlocks, blk, 0, stream>>>(ei, al, ar, mU);
  gat_edge_exp_kernel<<<eBlocks, blk, 0, stream>>>(ei, al, ar, mU, denom, ex);
  gat_edge_agg_kernel<<<aggBlocks, blk, 0, stream>>>(ei, ex, denom, xl, agg);
  gat_relu_bf_kernel<<<ncBlocks, blk, 0, stream>>>(agg, featb, NODES * HC);

  // ---- projections ----
  gat_gemm_kernel<HC, CH, 1><<<gemmBlocks, blk, 0, stream>>>(featb, wp1b, bp1, nullptr, p1b, NODES);
  gat_gemm_kernel<CH, CH, 2><<<gemmBlocks, blk, 0, stream>>>(p1b, wp2b, bp2, out, nullptr, NODES);
}
